// RepLKLiteBlock_6863357739681
// MI455X (gfx1250) — compile-verified, matrix operand order checked by reading, not run on hardware
//
#include <hip/hip_runtime.h>
#include <hip/hip_bf16.h>
#include <cstdint>

// ---------------------------------------------------------------------------
// RepLK-lite block for MI455X (gfx1250, wave32):
//   1) dwconv31:  31x31 depthwise conv. Plane staged to LDS by the Tensor
//                 Data Mover (tensor_load_to_lds, padded rows -> 96-f stride),
//                 weights via global_load_async_to_lds_b32 (ASYNCcnt path),
//                 4x4 register tile per lane, fused sum/sumsq reduction.
//   2) bn_finalize: per-channel scale/shift from stats.
//   3) pw_wmma:   1x1 conv as 256x256xPix GEMM, v_wmma_f32_16x16x32_f16,
//                 double-buffered LDS + register-pipelined global loads,
//                 BN+ReLU+f32->f16 fused into B-operand staging.
// ---------------------------------------------------------------------------

typedef __attribute__((ext_vector_type(16))) _Float16 v16h;
typedef __attribute__((ext_vector_type(8)))  _Float16 v8h;
typedef __attribute__((ext_vector_type(4)))  _Float16 v4h;
typedef __attribute__((ext_vector_type(8)))  float    v8f;
typedef __attribute__((ext_vector_type(4)))  uint32_t v4u;
typedef __attribute__((ext_vector_type(8)))  int      v8i;
typedef __attribute__((ext_vector_type(4)))  int      v4i;

#if defined(__has_builtin)
#if __has_builtin(__builtin_amdgcn_tensor_load_to_lds)
#define HAVE_TDM 1
#endif
#endif

#define BB   16
#define CC   256
#define HH   64
#define WW   64
#define KK   31
#define PADK 15
#define NPIX 4096                 // H*W per image
#define NTOT 65536                // B*H*W (BN normalization count)

// padded plane: (64+30) rows x stride 96; interior placed at column 16 so
// 16B-aligned destinations work. x col (w) lives at lds col w+16.
#define XROWS 94
#define XSTR  96
#define COFF  16

// ---------------------------------------------------------------------------
// Kernel 1: depthwise 31x31 conv + per-channel sum/sumsq partials
// grid = B*C blocks, 256 threads
// ---------------------------------------------------------------------------
__global__ void __launch_bounds__(256) dwconv31(const float* __restrict__ X,
                                                const float* __restrict__ DW,
                                                float* __restrict__ Y,
                                                float* __restrict__ stats) {
  __shared__ __attribute__((aligned(16))) float xs[XROWS * XSTR + 16];
  __shared__ __attribute__((aligned(16))) float wf[KK * KK];
  __shared__ float red[512];

  const int tid = threadIdx.x;
  const int wid = tid >> 5;
  const int bc  = blockIdx.x;          // b*C + c
  const int c   = bc & (CC - 1);

  // zero entire padded plane (covers halo), then DMA the 64x64 interior in
  for (int i = tid; i < XROWS * XSTR + 16; i += 256) xs[i] = 0.0f;
  __syncthreads();

  const float* xp = X + ((size_t)bc << 12);

#ifdef HAVE_TDM
  // ---- Tensor Data Mover: one descriptor moves the whole 64x64 f32 tile.
  // LDS padding: after every 64 DWORDs (pad_interval=5 -> 2^(5+1)) insert
  // 32 DWORDs (pad_amount=31 -> 31+1) so rows land on the 96-float stride.
  if (wid == 0) {
    const uint64_t ga = (uint64_t)(uintptr_t)xp;
    v4u g0;
    g0[0] = 1u;                                             // count=1
    g0[1] = (uint32_t)(uintptr_t)&xs[PADK * XSTR + COFF];   // lds_addr
    g0[2] = (uint32_t)ga;                                   // global_addr lo
    g0[3] = (uint32_t)((ga >> 32) & 0x01FFFFFFu) | 0x80000000u; // hi | type=2
    v8i g1;
    g1[0] = (int)0x3F520000u;  // pad_amount=31|pad_interval=5|pad_en|dsize=4B
    g1[1] = 64 << 16;          // tensor_dim0 = 64 (bits 79:48)
    g1[2] = 64 << 16;          // tensor_dim1 = 64 (bits 111:80)
    g1[3] = 64 << 16;          // tile_dim0   = 64 (bits 127:112)
    g1[4] = 64;                // tile_dim1   = 64, tile_dim2 = 0
    g1[5] = 64;                // tensor_dim0_stride = 64
    g1[6] = 0;
    g1[7] = 0;
    const v4i g2 = {0, 0, 0, 0};
    const v4i g3 = {0, 0, 0, 0};
    const v8i g4 = {0, 0, 0, 0, 0, 0, 0, 0};
    __builtin_amdgcn_tensor_load_to_lds(g0, g1, g2, g3, g4, 0);
#if __has_builtin(__builtin_amdgcn_s_wait_tensorcnt)
    __builtin_amdgcn_s_wait_tensorcnt(0);
#else
    asm volatile("s_wait_tensorcnt 0x0" ::: "memory");
#endif
  }
#else
  // ---- fallback: per-lane async DMA, 16B-aligned B128 into padded rows
#pragma unroll
  for (int i = 0; i < 4; ++i) {
    const int idx = tid + (i << 8);            // float4 index, 0..1023
    const int h = idx >> 4, w4 = idx & 15;     // 16 float4 per 64-wide row
    const uint32_t ldsb =
        (uint32_t)(uintptr_t)&xs[(h + PADK) * XSTR + COFF + (w4 << 2)];
    const uint64_t ga = (uint64_t)(uintptr_t)(xp + ((size_t)idx << 2));
    asm volatile("global_load_async_to_lds_b128 %0, %1, off"
                 :: "v"(ldsb), "v"(ga) : "memory");
  }
#endif

  // ---- weights: ASYNCcnt-tracked async copies into LDS
  for (int i = tid; i < KK * KK; i += 256) {
    const uint32_t ldsb = (uint32_t)(uintptr_t)&wf[i];
    const uint64_t ga = (uint64_t)(uintptr_t)(DW + (size_t)c * (KK * KK) + i);
    asm volatile("global_load_async_to_lds_b32 %0, %1, off"
                 :: "v"(ldsb), "v"(ga) : "memory");
  }
  asm volatile("s_wait_asynccnt 0x0" ::: "memory");
  __syncthreads();

  // each lane computes a 4x4 output tile
  const int ty = tid >> 4, tx = tid & 15;
  const int h0 = ty << 2, w0 = tx << 2;

  float tsum = 0.0f, tsq = 0.0f;
  float* yp = Y + ((size_t)bc << 12);

  for (int i = 0; i < 4; ++i) {
    const int h = h0 + i;
    float a0 = 0.0f, a1 = 0.0f, a2 = 0.0f, a3 = 0.0f;

    for (int kh = 0; kh < KK; ++kh) {
      const float* row = &xs[(h + kh) * XSTR + w0];   // 16B aligned
      float xr[36];
#pragma unroll
      for (int q = 0; q < 9; ++q) {
        const float4 t4 = *(const float4*)(row + 4 * q);
        xr[4 * q + 0] = t4.x;
        xr[4 * q + 1] = t4.y;
        xr[4 * q + 2] = t4.z;
        xr[4 * q + 3] = t4.w;
      }
      const float* wr = &wf[kh * KK];
      // out col w0+j needs x col w0+j+kw-15 -> lds col w0 + (kw+j+1)
#pragma unroll
      for (int kw = 0; kw < KK; ++kw) {
        const float wv = wr[kw];
        a0 = fmaf(wv, xr[kw + 1], a0);
        a1 = fmaf(wv, xr[kw + 2], a1);
        a2 = fmaf(wv, xr[kw + 3], a2);
        a3 = fmaf(wv, xr[kw + 4], a3);
      }
    }
    *(float4*)(yp + h * WW + w0) = make_float4(a0, a1, a2, a3);
    tsum += a0 + a1 + a2 + a3;
    tsq  += a0 * a0 + a1 * a1 + a2 * a2 + a3 * a3;
  }

  // block reduction of sum / sumsq
  red[tid]       = tsum;
  red[256 + tid] = tsq;
  __syncthreads();
#pragma unroll
  for (int s = 128; s > 0; s >>= 1) {
    if (tid < s) {
      red[tid]       += red[tid + s];
      red[256 + tid] += red[256 + tid + s];
    }
    __syncthreads();
  }
  if (tid == 0) {
    atomicAdd(&stats[c],       red[0]);
    atomicAdd(&stats[CC + c],  red[256]);
  }
}

// ---------------------------------------------------------------------------
// Kernel 2: finalize BN -> per-channel scale/shift
// ---------------------------------------------------------------------------
__global__ void __launch_bounds__(256) bn_finalize(const float* __restrict__ stats,
                                                   const float* __restrict__ gamma,
                                                   const float* __restrict__ beta,
                                                   float* __restrict__ ss) {
  const int c = threadIdx.x;
  const float invN = 1.0f / (float)NTOT;
  const float mean = stats[c] * invN;
  const float var  = stats[CC + c] * invN - mean * mean;
  const float scl  = gamma[c] * rsqrtf(var + 1e-5f);
  ss[c]      = scl;
  ss[CC + c] = beta[c] - mean * scl;
}

// ---------------------------------------------------------------------------
// Kernel 3: pointwise conv as WMMA GEMM, double-buffered and pipelined.
//   out[b,o,pix] = sum_c pw[o,c] * relu(scale[c]*y[b,c,pix]+shift[c])
// Block tile: M=128 (o) x N=64 (pix), K=256 in chunks of 32 (8 steps).
// 8 waves: (wm in 0..1) x (wn in 0..3); wave = 4 Mtiles x 1 Ntile.
// grid = (pixBlocks=64, oBlocks=2, b=16), 256 threads.
// ---------------------------------------------------------------------------
__global__ void __launch_bounds__(256) pw_wmma(const float* __restrict__ Y,
                                               const float* __restrict__ PW,
                                               const float* __restrict__ SS,
                                               float* __restrict__ OUT) {
  __shared__ __attribute__((aligned(16))) _Float16 ldsA[2][128 * 32]; // [o][k]
  __shared__ __attribute__((aligned(16))) _Float16 ldsB[2][64 * 32];  // [n][k]

  const int tid   = threadIdx.x;
  const int lane  = tid & 31;
  const int wid   = tid >> 5;
  const int wm    = wid & 1;       // M half (64 rows)
  const int wn    = wid >> 1;      // N tile (16 pixels)
  const int laneM = lane & 15;
  const int kHalf = (lane >> 4) << 3;  // 0 or 8 (per ISA 16-bit A/B layout)

  const int pixBase = blockIdx.x << 6;   // 64-pixel block
  const int oBase   = blockIdx.y << 7;   // 128-row block
  const int b       = blockIdx.z;

  // register staging for the in-flight chunk
  float4 ra[4];
  float4 rb[2];
  float  rscl[2], rsh[2];

  auto load_chunk = [&](int kc) {
#pragma unroll
    for (int i = 0; i < 4; ++i) {
      const int idx = tid + (i << 8);        // float4 index, 0..1023
      const int o = idx >> 3, k4 = (idx & 7) << 2;
      ra[i] = *(const float4*)&PW[(size_t)(oBase + o) * CC + kc + k4];
    }
#pragma unroll
    for (int i = 0; i < 2; ++i) {
      const int idx = tid + (i << 8);        // float4 index, 0..511
      const int cl = idx >> 4, n4 = (idx & 15) << 2;
      rscl[i] = SS[kc + cl];
      rsh[i]  = SS[CC + kc + cl];
      const size_t ga = ((size_t)(b * CC + kc + cl) << 12) + pixBase + n4;
      rb[i] = *(const float4*)&Y[ga];
      if (kc + 32 < CC)  // warm the chunk after next (global_prefetch_b8)
        __builtin_prefetch(&Y[ga + ((size_t)32 << 12)], 0, 3);
    }
  };

  auto store_chunk = [&](int buf) {
#pragma unroll
    for (int i = 0; i < 4; ++i) {
      const int idx = tid + (i << 8);
      const int o = idx >> 3, k4 = (idx & 7) << 2;
      v4h h4;
      h4[0] = (_Float16)ra[i].x; h4[1] = (_Float16)ra[i].y;
      h4[2] = (_Float16)ra[i].z; h4[3] = (_Float16)ra[i].w;
      *(v4h*)&ldsA[buf][(o << 5) + k4] = h4;   // 8B LDS store
    }
#pragma unroll
    for (int i = 0; i < 2; ++i) {
      const int idx = tid + (i << 8);
      const int cl = idx >> 4, n4 = (idx & 15) << 2;
      const float r0 = fmaxf(fmaf(rb[i].x, rscl[i], rsh[i]), 0.0f);
      const float r1 = fmaxf(fmaf(rb[i].y, rscl[i], rsh[i]), 0.0f);
      const float r2 = fmaxf(fmaf(rb[i].z, rscl[i], rsh[i]), 0.0f);
      const float r3 = fmaxf(fmaf(rb[i].w, rscl[i], rsh[i]), 0.0f);
      ldsB[buf][((n4 + 0) << 5) + cl] = (_Float16)r0;   // transposed [n][k]
      ldsB[buf][((n4 + 1) << 5) + cl] = (_Float16)r1;
      ldsB[buf][((n4 + 2) << 5) + cl] = (_Float16)r2;
      ldsB[buf][((n4 + 3) << 5) + cl] = (_Float16)r3;
    }
  };

  load_chunk(0);
  store_chunk(0);
  __syncthreads();

  v8f acc[4] = {};

#pragma unroll
  for (int k = 0; k < 8; ++k) {
    if (k < 7) load_chunk((k + 1) << 5);     // global loads fly over compute

    const int buf = k & 1;
    // ---- B fragment (32x16): lane holds column n, two 8-half k runs
    const int nrow = ((wn << 4) + laneM) << 5;
    const v8h blo = *(const v8h*)&ldsB[buf][nrow + kHalf];
    const v8h bhi = *(const v8h*)&ldsB[buf][nrow + 16 + kHalf];
    const v16h bf = __builtin_shufflevector(blo, bhi, 0, 1, 2, 3, 4, 5, 6, 7,
                                            8, 9, 10, 11, 12, 13, 14, 15);
    // ---- 4 A fragments + 4 WMMAs
#pragma unroll
    for (int t = 0; t < 4; ++t) {
      const int orow = ((wm << 6) + (t << 4) + laneM) << 5;
      const v8h alo = *(const v8h*)&ldsA[buf][orow + kHalf];
      const v8h ahi = *(const v8h*)&ldsA[buf][orow + 16 + kHalf];
      const v16h af = __builtin_shufflevector(alo, ahi, 0, 1, 2, 3, 4, 5, 6, 7,
                                              8, 9, 10, 11, 12, 13, 14, 15);
      acc[t] = __builtin_amdgcn_wmma_f32_16x16x32_f16(
          false, af, false, bf, (short)0, acc[t], false, false);
    }

    if (k < 7) store_chunk((k + 1) & 1);     // fill the other buffer
    __syncthreads();
  }

  // ---- store C/D (VGPR j -> M = j + 8*(lane/16); N = lane%16)
  const int mAdd = (lane >> 4) << 3;
  const int pix  = pixBase + (wn << 4) + laneM;
#pragma unroll
  for (int t = 0; t < 4; ++t) {
#pragma unroll
    for (int j = 0; j < 8; ++j) {
      const int o = oBase + (wm << 6) + (t << 4) + j + mAdd;
      OUT[((size_t)(b * CC + o) << 12) + pix] = acc[t][j];
    }
  }
}

// ---------------------------------------------------------------------------
// Launch. Inputs: x, dw_w, gamma, beta, pw_w (all f32). Output f32.
// ws layout (floats): [0,256)=sum  [256,512)=sumsq  [512,1024)=scale/shift
//                     [1024, 1024+16777216) = y buffer
// ---------------------------------------------------------------------------
extern "C" void kernel_launch(void* const* d_in, const int* in_sizes, int n_in,
                              void* d_out, int out_size, void* d_ws, size_t ws_size,
                              hipStream_t stream) {
  const float* x     = (const float*)d_in[0];
  const float* dw_w  = (const float*)d_in[1];
  const float* gamma = (const float*)d_in[2];
  const float* beta  = (const float*)d_in[3];
  const float* pw_w  = (const float*)d_in[4];
  float*       out   = (float*)d_out;

  float* stats = (float*)d_ws;           // 512 floats
  float* ss    = stats + 512;            // 512 floats
  float* ybuf  = stats + 1024;           // B*C*H*W floats

  hipMemsetAsync(stats, 0, 512 * sizeof(float), stream);

  dwconv31<<<BB * CC, 256, 0, stream>>>(x, dw_w, ybuf, stats);
  bn_finalize<<<1, 256, 0, stream>>>(stats, gamma, beta, ss);
  pw_wmma<<<dim3(NPIX / 64, CC / 128, BB), 256, 0, stream>>>(ybuf, pw_w, ss, out);
}
